// transformer_74045236183283
// MI455X (gfx1250) — matrix operand order repaired and checked by reading, not compiled, not run on hardware
//
#include <hip/hip_runtime.h>
#include <hip/hip_bf16.h>

// ---------------------------------------------------------------------------
// Channel-attention transformer block for MI455X (gfx1250, wave32, WMMA).
//   q,k,v = W{q,k,v} @ x  ->  attn = softmax(q.kT/sqrt(512))  ->  out = attn@v
// All contractions run on v_wmma_f32_16x16x32_bf16 (fp32 accumulate).
// Double-buffered LDS + register/async prefetch pipelines each GEMM.
// ---------------------------------------------------------------------------

#define BATCH 16
#define DIM   512
#define NPTS  4096

typedef __bf16 bf16_t;
typedef __attribute__((ext_vector_type(16))) __bf16 v16bf;
typedef __attribute__((ext_vector_type(8)))  __bf16 v8bf;
typedef __attribute__((ext_vector_type(4)))  __bf16 v4bf;
typedef __attribute__((ext_vector_type(8)))  float  v8f;

// ---- gfx1250 async global->LDS copy (guarded; sync fallback) ---------------
#if defined(__has_builtin)
#if __has_builtin(__builtin_amdgcn_global_load_async_to_lds_b128)
#define USE_ASYNC 1
#endif
#endif
#ifndef USE_ASYNC
#define USE_ASYNC 0
#endif

#if USE_ASYNC
typedef int v4i_t __attribute__((ext_vector_type(4)));
typedef __attribute__((address_space(1))) v4i_t* gv4i_p;   // global int4*
typedef __attribute__((address_space(3))) v4i_t* lv4i_p;   // LDS int4*
__device__ __forceinline__ void copy_b128(bf16_t* lds_dst, const bf16_t* gsrc) {
  // 16B DMA directly into LDS, tracked by ASYNCcnt (no VGPR round-trip)
  __builtin_amdgcn_global_load_async_to_lds_b128(
      (gv4i_p)gsrc, (lv4i_p)lds_dst, 0, 0);
}
#if __has_builtin(__builtin_amdgcn_s_wait_asynccnt)
__device__ __forceinline__ void wait_async0() { __builtin_amdgcn_s_wait_asynccnt(0); }
#else
__device__ __forceinline__ void wait_async0() {
  asm volatile("s_wait_asynccnt 0x0" ::: "memory");
}
#endif
#else
__device__ __forceinline__ void copy_b128(bf16_t* lds_dst, const bf16_t* gsrc) {
  *reinterpret_cast<uint4*>(lds_dst) = *reinterpret_cast<const uint4*>(gsrc);
}
__device__ __forceinline__ void wait_async0() {}
#endif

// ---- WMMA helper -----------------------------------------------------------
__device__ __forceinline__ v8f wmma_bf16(v16bf a, v16bf b, v8f c) {
  return __builtin_amdgcn_wmma_f32_16x16x32_bf16(
      /*neg_a=*/false, a, /*neg_b=*/false, b,
      /*c_mod=*/(short)0, c, /*reuse_a=*/false, /*reuse_b=*/false);
}

// ---- Fragment loads from LDS ----------------------------------------------
// A tile in LDS: [rows][32] bf16 row-major.  16-bit A 16x32 layout:
//   lanes 0-15 : row M=lane,    K = 0..7 then 16..23
//   lanes 16-31: row M=lane-16, K = 8..15 then 24..31
__device__ __forceinline__ v16bf load_frag_A(const bf16_t* tile, int mBase, int lane) {
  const int r  = mBase + (lane & 15);
  const int hi = (lane >> 4) * 8;
  const bf16_t* p = tile + r * 32 + hi;
  v16bf a;
  uint4* u = reinterpret_cast<uint4*>(&a);
  u[0] = *reinterpret_cast<const uint4*>(p);
  u[1] = *reinterpret_cast<const uint4*>(p + 16);
  return a;
}

// B tile in LDS: [cols][32] bf16 (column n, then K).  16-bit B 32x16 layout:
//   lanes 0-15 : col N=lane,    K = 0..7 then 16..23 ; lanes 16-31: col N=lane-16, K = 8..15 then 24..31
__device__ __forceinline__ v16bf load_frag_B(const bf16_t* tile, int nBase, int lane) {
  const int c  = nBase + (lane & 15);
  const int hi = (lane >> 4) * 8;
  const bf16_t* p = tile + c * 32 + hi;
  v16bf b;
  uint4* u = reinterpret_cast<uint4*>(&b);
  u[0] = *reinterpret_cast<const uint4*>(p);
  u[1] = *reinterpret_cast<const uint4*>(p + 16);
  return b;
}

// ---- C/D tile stores (lane=N, vgpr i = row m0 + (lane>=16?8:0)+i) ----------
__device__ __forceinline__ void store_tile_f32(float* out, int ld, int m0, int n0,
                                               v8f c, int lane, float scale) {
  const int n = n0 + (lane & 15);
  const int m = m0 + (lane >> 4) * 8;
#pragma unroll
  for (int i = 0; i < 8; ++i)
    out[(size_t)(m + i) * ld + n] = c[i] * scale;
}

__device__ __forceinline__ void store_tile_bf16(bf16_t* out, int ld, int m0, int n0,
                                                v8f c, int lane) {
  const int n = n0 + (lane & 15);
  const int m = m0 + (lane >> 4) * 8;
#pragma unroll
  for (int i = 0; i < 8; ++i)
    out[(size_t)(m + i) * ld + n] = (bf16_t)c[i];
}

// ---------------------------------------------------------------------------
// Stage 1: fused QKV. {q,k,v}[d,n] = W{q,k,v}[d,c] * x[c,n] per batch, bf16 out.
// Register-prefetch pipeline (fp32->bf16 convert + x transpose happen in regs).
// Grid: (N/64, DIM/128, B); block 256 = 8 waves (4x2), wave tile 32x32.
// ---------------------------------------------------------------------------
struct QkvRegs {
  float4 a[3][4];  // 3 W matrices x 4 row-passes of float4
  v8bf   b;        // 8 strided x values (transposed column), converted
};

__device__ __forceinline__ void qkv_load_regs(
    QkvRegs& r, const float* __restrict__ Wq, const float* __restrict__ Wk,
    const float* __restrict__ Wv, const float* __restrict__ xb,
    int d0, int k0, int n0, int tid) {
  const float* const Ws[3] = { Wq, Wk, Wv };
  const int ar = tid >> 3;
  const int c4 = (tid & 7) * 4;
#pragma unroll
  for (int w = 0; w < 3; ++w)
#pragma unroll
    for (int p = 0; p < 4; ++p)
      r.a[w][p] = *reinterpret_cast<const float4*>(
          Ws[w] + (size_t)(d0 + ar + p * 32) * DIM + k0 + c4);
  const int n  = tid & 63;
  const int kb = (tid >> 6) * 8;
#pragma unroll
  for (int i = 0; i < 8; ++i)
    r.b[i] = (bf16_t)xb[(size_t)(k0 + kb + i) * NPTS + n0 + n];  // coalesced
}

__device__ __forceinline__ void qkv_store_lds(
    bf16_t (* __restrict__ tA)[128 * 32], bf16_t* __restrict__ tB,
    const QkvRegs& r, int tid) {
  const int ar = tid >> 3;
  const int c4 = (tid & 7) * 4;
#pragma unroll
  for (int w = 0; w < 3; ++w)
#pragma unroll
    for (int p = 0; p < 4; ++p) {
      const float4 f = r.a[w][p];
      v4bf h = { (bf16_t)f.x, (bf16_t)f.y, (bf16_t)f.z, (bf16_t)f.w };
      *reinterpret_cast<v4bf*>(tA[w] + (ar + p * 32) * 32 + c4) = h;
    }
  const int n  = tid & 63;
  const int kb = (tid >> 6) * 8;
  *reinterpret_cast<v8bf*>(tB + n * 32 + kb) = r.b;  // ds_store_b128
}

__global__ __launch_bounds__(256) void qkv_gemm_kernel(
    const float* __restrict__ x, const float* __restrict__ Wq,
    const float* __restrict__ Wk, const float* __restrict__ Wv,
    bf16_t* __restrict__ q, bf16_t* __restrict__ k, bf16_t* __restrict__ v) {
  __shared__ bf16_t tA[2][3][128 * 32];   // 48KB
  __shared__ bf16_t tB[2][64 * 32];       // 8KB
  const int tid  = threadIdx.x;
  const int lane = tid & 31;
  const int wave = tid >> 5;
  const int wm   = (wave & 3) * 32;
  const int wn   = (wave >> 2) * 32;
  const int n0   = blockIdx.x * 64;
  const int d0   = blockIdx.y * 128;
  const int b    = blockIdx.z;
  const float* xb = x + (size_t)b * DIM * NPTS;

  v8f acc[3][2][2] = {};
  QkvRegs r;

  qkv_load_regs(r, Wq, Wk, Wv, xb, d0, 0, n0, tid);
  qkv_store_lds(tA[0], tB[0], r, tid);
  __syncthreads();

  for (int it = 0; it < DIM / 32; ++it) {
    const int buf = it & 1;
    if (it + 1 < DIM / 32)
      qkv_load_regs(r, Wq, Wk, Wv, xb, d0, (it + 1) * 32, n0, tid);

    v16bf bf0 = load_frag_B(tB[buf], wn + 0,  lane);
    v16bf bf1 = load_frag_B(tB[buf], wn + 16, lane);
#pragma unroll
    for (int m = 0; m < 2; ++m) {
#pragma unroll
      for (int w = 0; w < 3; ++w) {
        v16bf af = load_frag_A(tA[buf][w], wm + m * 16, lane);
        acc[w][m][0] = wmma_bf16(af, bf0, acc[w][m][0]);
        acc[w][m][1] = wmma_bf16(af, bf1, acc[w][m][1]);
      }
    }
    if (it + 1 < DIM / 32) qkv_store_lds(tA[buf ^ 1], tB[buf ^ 1], r, tid);
    __syncthreads();
  }

  bf16_t* const outs[3] = { q + (size_t)b * DIM * NPTS,
                            k + (size_t)b * DIM * NPTS,
                            v + (size_t)b * DIM * NPTS };
#pragma unroll
  for (int w = 0; w < 3; ++w)
#pragma unroll
    for (int m = 0; m < 2; ++m)
#pragma unroll
      for (int nn = 0; nn < 2; ++nn)
        store_tile_bf16(outs[w], NPTS, d0 + wm + m * 16, n0 + wn + nn * 16,
                        acc[w][m][nn], lane);
}

// ---------------------------------------------------------------------------
// Stage 2: logits[d,e] = (1/sqrt(DIM)) * sum_n q[d,n]*k[e,n].  fp32 out.
// Both operands are straight bf16 row copies -> async global->LDS DMA.
// Grid: (DIM/64, DIM/128, B).
// ---------------------------------------------------------------------------
__device__ __forceinline__ void qk_issue_tiles(
    bf16_t* __restrict__ dA, bf16_t* __restrict__ dB,
    const bf16_t* __restrict__ qb, const bf16_t* __restrict__ kb,
    int d0, int e0, int nk, int tid) {
#pragma unroll
  for (int p = 0; p < 2; ++p) {             // A: 128x32 bf16 = 512 x 16B chunks
    const int idx = tid + p * 256;
    const int rr  = idx >> 2;
    const int c8  = (idx & 3) * 8;
    copy_b128(dA + rr * 32 + c8, qb + (size_t)(d0 + rr) * NPTS + nk + c8);
  }
  {                                          // B: 64x32 bf16 = 256 x 16B chunks
    const int rr = tid >> 2;
    const int c8 = (tid & 3) * 8;
    copy_b128(dB + rr * 32 + c8, kb + (size_t)(e0 + rr) * NPTS + nk + c8);
  }
}

__global__ __launch_bounds__(256) void qk_gemm_kernel(
    const bf16_t* __restrict__ q, const bf16_t* __restrict__ k,
    float* __restrict__ logits) {
  __shared__ bf16_t tA[2][128 * 32];
  __shared__ bf16_t tB[2][64 * 32];
  const int tid  = threadIdx.x;
  const int lane = tid & 31;
  const int wave = tid >> 5;
  const int wm   = (wave & 3) * 32;
  const int wn   = (wave >> 2) * 32;
  const int e0   = blockIdx.x * 64;
  const int d0   = blockIdx.y * 128;
  const int b    = blockIdx.z;
  const bf16_t* qb = q + (size_t)b * DIM * NPTS;
  const bf16_t* kb = k + (size_t)b * DIM * NPTS;

  v8f acc[2][2] = {};

  qk_issue_tiles(tA[0], tB[0], qb, kb, d0, e0, 0, tid);
  wait_async0();
  __syncthreads();

  for (int it = 0; it < NPTS / 32; ++it) {
    const int buf = it & 1;
    if (it + 1 < NPTS / 32)
      qk_issue_tiles(tA[buf ^ 1], tB[buf ^ 1], qb, kb, d0, e0, (it + 1) * 32, tid);

    v16bf bf0 = load_frag_B(tB[buf], wn + 0,  lane);
    v16bf bf1 = load_frag_B(tB[buf], wn + 16, lane);
#pragma unroll
    for (int m = 0; m < 2; ++m) {
      v16bf af = load_frag_A(tA[buf], wm + m * 16, lane);
      acc[m][0] = wmma_bf16(af, bf0, acc[m][0]);
      acc[m][1] = wmma_bf16(af, bf1, acc[m][1]);
    }
    wait_async0();       // our DMA for next buffer has landed
    __syncthreads();     // everyone's has; prev readers of next buffer done
  }

  const float inv_temp = 0.04419417382415922f;  // 1/sqrt(512)
  float* lb = logits + (size_t)b * DIM * DIM;
#pragma unroll
  for (int m = 0; m < 2; ++m)
#pragma unroll
    for (int nn = 0; nn < 2; ++nn)
      store_tile_f32(lb, DIM, d0 + wm + m * 16, e0 + wn + nn * 16,
                     acc[m][nn], lane, inv_temp);
}

// ---------------------------------------------------------------------------
// Stage 3: row softmax over 512-wide rows; one wave32 per row; bf16 out.
// ---------------------------------------------------------------------------
__global__ __launch_bounds__(256) void softmax_kernel(
    const float* __restrict__ logits, bf16_t* __restrict__ attn) {
  const int row  = blockIdx.x * 8 + (threadIdx.x >> 5);
  const int lane = threadIdx.x & 31;
  const float* rp = logits + (size_t)row * DIM;

  float vals[16];
  float mx = -3.4e38f;
#pragma unroll
  for (int i = 0; i < 16; ++i) {
    vals[i] = rp[lane + i * 32];
    mx = fmaxf(mx, vals[i]);
  }
#pragma unroll
  for (int off = 16; off > 0; off >>= 1)
    mx = fmaxf(mx, __shfl_xor(mx, off, 32));

  float sum = 0.f;
#pragma unroll
  for (int i = 0; i < 16; ++i) {
    vals[i] = __expf(vals[i] - mx);
    sum += vals[i];
  }
#pragma unroll
  for (int off = 16; off > 0; off >>= 1)
    sum += __shfl_xor(sum, off, 32);

  const float inv = 1.0f / sum;
  bf16_t* op = attn + (size_t)row * DIM;
#pragma unroll
  for (int i = 0; i < 16; ++i)
    op[lane + i * 32] = (bf16_t)(vals[i] * inv);
}

// ---------------------------------------------------------------------------
// Stage 4: out[d,n] = sum_e attn[d,e] * v[e,n].  fp32 out to d_out.
// A (attn) is a straight copy -> async DMA; B (v) needs transpose -> regs.
// Grid: (N/64, DIM/128, B).
// ---------------------------------------------------------------------------
__device__ __forceinline__ void av_issue_A(
    bf16_t* __restrict__ dA, const bf16_t* __restrict__ ab,
    int d0, int k0, int tid) {
#pragma unroll
  for (int p = 0; p < 2; ++p) {
    const int idx = tid + p * 256;
    const int rr  = idx >> 2;
    const int c8  = (idx & 3) * 8;
    copy_b128(dA + rr * 32 + c8, ab + (size_t)(d0 + rr) * DIM + k0 + c8);
  }
}

__device__ __forceinline__ v8bf av_load_B(const bf16_t* __restrict__ vb,
                                          int k0, int n0, int tid) {
  const int n  = tid & 63;
  const int kb = (tid >> 6) * 8;
  v8bf t;
#pragma unroll
  for (int i = 0; i < 8; ++i)
    t[i] = vb[(size_t)(k0 + kb + i) * NPTS + n0 + n];
  return t;
}

__device__ __forceinline__ void av_store_B(bf16_t* __restrict__ dB, v8bf t, int tid) {
  const int n  = tid & 63;
  const int kb = (tid >> 6) * 8;
  *reinterpret_cast<v8bf*>(dB + n * 32 + kb) = t;
}

__global__ __launch_bounds__(256) void av_gemm_kernel(
    const bf16_t* __restrict__ attn, const bf16_t* __restrict__ v,
    float* __restrict__ out) {
  __shared__ bf16_t tA[2][128 * 32];
  __shared__ bf16_t tB[2][64 * 32];
  const int tid  = threadIdx.x;
  const int lane = tid & 31;
  const int wave = tid >> 5;
  const int wm   = (wave & 3) * 32;
  const int wn   = (wave >> 2) * 32;
  const int n0   = blockIdx.x * 64;
  const int d0   = blockIdx.y * 128;
  const int b    = blockIdx.z;
  const bf16_t* ab = attn + (size_t)b * DIM * DIM;
  const bf16_t* vb = v    + (size_t)b * DIM * NPTS;

  v8f acc[2][2] = {};

  av_issue_A(tA[0], ab, d0, 0, tid);
  av_store_B(tB[0], av_load_B(vb, 0, n0, tid), tid);
  wait_async0();
  __syncthreads();

  v8bf rb;
  for (int it = 0; it < DIM / 32; ++it) {
    const int buf = it & 1;
    if (it + 1 < DIM / 32) {
      av_issue_A(tA[buf ^ 1], ab, d0, (it + 1) * 32, tid);
      rb = av_load_B(vb, (it + 1) * 32, n0, tid);
    }

    v16bf bf0 = load_frag_B(tB[buf], wn + 0,  lane);
    v16bf bf1 = load_frag_B(tB[buf], wn + 16, lane);
#pragma unroll
    for (int m = 0; m < 2; ++m) {
      v16bf af = load_frag_A(tA[buf], wm + m * 16, lane);
      acc[m][0] = wmma_bf16(af, bf0, acc[m][0]);
      acc[m][1] = wmma_bf16(af, bf1, acc[m][1]);
    }
    if (it + 1 < DIM / 32) av_store_B(tB[buf ^ 1], rb, tid);
    wait_async0();
    __syncthreads();
  }

  float* ob = out + (size_t)b * DIM * NPTS;
#pragma unroll
  for (int m = 0; m < 2; ++m)
#pragma unroll
    for (int nn = 0; nn < 2; ++nn)
      store_tile_f32(ob, NPTS, d0 + wm + m * 16, n0 + wn + nn * 16,
                     acc[m][nn], lane, 1.0f);
}

// ---------------------------------------------------------------------------
// Workspace layout (bytes):
//   q     bf16 @ 0          (67108864)   -- dead after stage 2; attn reuses it
//   k     bf16 @ 67108864   (67108864)
//   v     bf16 @ 134217728  (67108864)
//   logit f32  @ 201326592  (16777216)
//   attn  bf16 @ 0          (8388608, overwrites dead q)
// Total required: 218103808 bytes.
// ---------------------------------------------------------------------------
extern "C" void kernel_launch(void* const* d_in, const int* in_sizes, int n_in,
                              void* d_out, int out_size, void* d_ws, size_t ws_size,
                              hipStream_t stream) {
  const float* x  = (const float*)d_in[0];
  const float* Wq = (const float*)d_in[1];
  const float* Wk = (const float*)d_in[2];
  const float* Wv = (const float*)d_in[3];
  float* out = (float*)d_out;

  char* ws = (char*)d_ws;
  bf16_t* q      = (bf16_t*)(ws + 0);
  bf16_t* k      = (bf16_t*)(ws + 67108864);
  bf16_t* v      = (bf16_t*)(ws + 134217728);
  float*  logits = (float*) (ws + 201326592);
  bf16_t* attn   = (bf16_t*)(ws + 0);          // reuse q region

  dim3 blk(256);
  qkv_gemm_kernel<<<dim3(NPTS / 64, DIM / 128, BATCH), blk, 0, stream>>>(
      x, Wq, Wk, Wv, q, k, v);
  qk_gemm_kernel<<<dim3(DIM / 64, DIM / 128, BATCH), blk, 0, stream>>>(
      q, k, logits);
  softmax_kernel<<<dim3(BATCH * DIM / 8), blk, 0, stream>>>(logits, attn);
  av_gemm_kernel<<<dim3(NPTS / 64, DIM / 128, BATCH), blk, 0, stream>>>(
      attn, v, out);
}
